// SVGDModel_70531952935585
// MI455X (gfx1250) — matrix operand order repaired and checked by reading, not compiled, not run on hardware
//
#include <hip/hip_runtime.h>

#define N 4096
#define D 128

typedef __attribute__((ext_vector_type(16))) __bf16 v16bf;
typedef __attribute__((ext_vector_type(8)))  __bf16 v8bf;
typedef __attribute__((ext_vector_type(8)))  float  v8f;

union BF16x16 { v16bf v; v8bf h[2]; };

// ---------------- prep: split P into bf16 hi/lo, compute row norms ----------------
__global__ void svgd_prep(const float* __restrict__ P, float* __restrict__ sq,
                          __bf16* __restrict__ Phi, __bf16* __restrict__ Plo) {
    int i = blockIdx.x;          // row
    int d = threadIdx.x;         // 0..127
    float x = P[i * D + d];
    __bf16 h = (__bf16)x;
    __bf16 l = (__bf16)(x - (float)h);
    Phi[i * D + d] = h;
    Plo[i * D + d] = l;

    float s = x * x;
    for (int off = 16; off > 0; off >>= 1) s += __shfl_down(s, off, 32);
    __shared__ float ps[4];
    int lane = threadIdx.x & 31, wv = threadIdx.x >> 5;
    if (lane == 0) ps[wv] = s;
    __syncthreads();
    if (threadIdx.x == 0) sq[i] = ps[0] + ps[1] + ps[2] + ps[3];
}

// ---------------- prep2: 32x32 LDS-tiled transpose of the bf16 splits ----------------
__global__ void svgd_transpose(const __bf16* __restrict__ Phi, const __bf16* __restrict__ Plo,
                               __bf16* __restrict__ PThi, __bf16* __restrict__ PTlo) {
    __shared__ __bf16 th[32][33];
    __shared__ __bf16 tl[32][33];
    int bi = blockIdx.x * 32;    // i base
    int bd = blockIdx.y * 32;    // d base
    int x = threadIdx.x;         // 0..31
    for (int yy = threadIdx.y; yy < 32; yy += 8) {
        th[yy][x] = Phi[(bi + yy) * D + bd + x];
        tl[yy][x] = Plo[(bi + yy) * D + bd + x];
    }
    __syncthreads();
    for (int yy = threadIdx.y; yy < 32; yy += 8) {
        PThi[(bd + yy) * N + bi + x] = th[x][yy];
        PTlo[(bd + yy) * N + bi + x] = tl[x][yy];
    }
}

// ---------------- fused SVGD main kernel ----------------
// block: 256 threads = 8 waves. Each block owns 16 output rows (ibase..ibase+15).
// Wave w processes j in [w*512, (w+1)*512), accumulating GP (16x128 fp32) and row sums.
__global__ __launch_bounds__(256)
void svgd_main(const float* __restrict__ particles, const float* __restrict__ mu,
               const float* __restrict__ bw, const float* __restrict__ sq,
               const __bf16* __restrict__ Phi, const __bf16* __restrict__ Plo,
               const __bf16* __restrict__ PThi, const __bf16* __restrict__ PTlo,
               float* __restrict__ out) {
    const int tid  = threadIdx.x;
    const int w    = tid >> 5;
    const int lane = tid & 31;
    const int m    = lane & 15;   // row/col-in-tile index
    const int hf   = lane >> 4;   // lane half
    const int ibase = blockIdx.x * 16;

    __shared__ float lds_g[8][16 * 36];   // per-wave gram tile, padded stride 36
    __shared__ float accS[16 * 128];      // block GP accumulator
    __shared__ float rsS[16];             // block row sums

    for (int k = tid; k < 16 * 128; k += 256) accS[k] = 0.0f;
    if (tid < 16) rsS[tid] = 0.0f;
    __syncthreads();

    const float h2     = bw[0] * bw[0];
    const float inv2h2 = 0.5f / h2;
    const float invh2  = 1.0f / h2;
    const float invN   = 1.0f / (float)N;

    // ---- preload A fragments (this block's 16 rows, K=D=128 -> 4 chunks of 32) ----
    BF16x16 Ahi[4], Alo[4];
    {
        const __bf16* rh = Phi + (ibase + m) * D;
        const __bf16* rl = Plo + (ibase + m) * D;
#pragma unroll
        for (int kc = 0; kc < 4; ++kc) {
            Ahi[kc].h[0] = *(const v8bf*)(rh + kc * 32 + 8 * hf);
            Ahi[kc].h[1] = *(const v8bf*)(rh + kc * 32 + 16 + 8 * hf);
            Alo[kc].h[0] = *(const v8bf*)(rl + kc * 32 + 8 * hf);
            Alo[kc].h[1] = *(const v8bf*)(rl + kc * 32 + 16 + 8 * hf);
        }
    }
    float sqi[8];
#pragma unroll
    for (int r = 0; r < 8; ++r) sqi[r] = sq[ibase + r + 8 * hf];

    const v8f vzero = {0.f, 0.f, 0.f, 0.f, 0.f, 0.f, 0.f, 0.f};
    v8f acc[8];
#pragma unroll
    for (int dc = 0; dc < 8; ++dc) acc[dc] = vzero;
    float rs = 0.0f;

    const int jbeg = w * 512, jend = jbeg + 512;
    for (int jb = jbeg; jb < jend; jb += 32) {
        // ---- GEMM1: 16x32 distance dot tile, bf16 hi/lo compensated (fp32-accurate) ----
        v8f dot[2];
        dot[0] = vzero; dot[1] = vzero;
#pragma unroll
        for (int t = 0; t < 2; ++t) {
            const __bf16* bh = Phi + (jb + 16 * t + m) * D + 8 * hf;
            const __bf16* bl = Plo + (jb + 16 * t + m) * D + 8 * hf;
#pragma unroll
            for (int kc = 0; kc < 4; ++kc) {
                BF16x16 Bh, Bl;
                Bh.h[0] = *(const v8bf*)(bh + kc * 32);
                Bh.h[1] = *(const v8bf*)(bh + kc * 32 + 16);
                Bl.h[0] = *(const v8bf*)(bl + kc * 32);
                Bl.h[1] = *(const v8bf*)(bl + kc * 32 + 16);
                dot[t] = __builtin_amdgcn_wmma_f32_16x16x32_bf16(false, Ahi[kc].v, false, Bh.v,
                                                                 (short)0, dot[t], false, false);
                dot[t] = __builtin_amdgcn_wmma_f32_16x16x32_bf16(false, Ahi[kc].v, false, Bl.v,
                                                                 (short)0, dot[t], false, false);
                dot[t] = __builtin_amdgcn_wmma_f32_16x16x32_bf16(false, Alo[kc].v, false, Bh.v,
                                                                 (short)0, dot[t], false, false);
            }
        }

        // ---- exp() and park gram tile in LDS (C-layout -> padded row-major) ----
#pragma unroll
        for (int t = 0; t < 2; ++t) {
            float sqj = sq[jb + 16 * t + m];
#pragma unroll
            for (int r = 0; r < 8; ++r) {
                float d2 = sqi[r] + sqj - 2.0f * dot[t][r];
                float g  = __expf(-d2 * inv2h2);
                lds_g[w][(r + 8 * hf) * 36 + 16 * t + m] = g;
            }
        }
        __syncthreads();

        // ---- re-read gram as A-fragment (hi/lo) + partial row sums ----
        BF16x16 Ghi, Glo;
        {
            const float* gr = &lds_g[w][m * 36];
#pragma unroll
            for (int e = 0; e < 8; ++e) {
                float g0 = gr[8 * hf + e];        // K = e + 8*hf
                float g1 = gr[16 + 8 * hf + e];   // K = e + 16 + 8*hf
                rs += g0 + g1;
                __bf16 h0 = (__bf16)g0;
                __bf16 h1 = (__bf16)g1;
                Ghi.v[e]     = h0;  Glo.v[e]     = (__bf16)(g0 - (float)h0);
                Ghi.v[e + 8] = h1;  Glo.v[e + 8] = (__bf16)(g1 - (float)h1);
            }
        }

        // ---- GEMM2: GP += gram @ P   (B from transposed bf16 splits) ----
#pragma unroll
        for (int dc = 0; dc < 8; ++dc) {
            const __bf16* bth = PThi + (dc * 16 + m) * N + jb + 8 * hf;
            const __bf16* btl = PTlo + (dc * 16 + m) * N + jb + 8 * hf;
            BF16x16 Bh, Bl;
            Bh.h[0] = *(const v8bf*)(bth);
            Bh.h[1] = *(const v8bf*)(bth + 16);
            Bl.h[0] = *(const v8bf*)(btl);
            Bl.h[1] = *(const v8bf*)(btl + 16);
            acc[dc] = __builtin_amdgcn_wmma_f32_16x16x32_bf16(false, Ghi.v, false, Bh.v,
                                                              (short)0, acc[dc], false, false);
            acc[dc] = __builtin_amdgcn_wmma_f32_16x16x32_bf16(false, Ghi.v, false, Bl.v,
                                                              (short)0, acc[dc], false, false);
            acc[dc] = __builtin_amdgcn_wmma_f32_16x16x32_bf16(false, Glo.v, false, Bh.v,
                                                              (short)0, acc[dc], false, false);
        }
        __syncthreads();
    }

    // ---- merge wave partials into block accumulators ----
    rs += __shfl_xor(rs, 16, 32);          // both halves of row m combined
    if (hf == 0) atomicAdd(&rsS[m], rs);
#pragma unroll
    for (int dc = 0; dc < 8; ++dc)
#pragma unroll
        for (int r = 0; r < 8; ++r)
            atomicAdd(&accS[(r + 8 * hf) * 128 + dc * 16 + m], acc[dc][r]);
    __syncthreads();

    // ---- SVGD epilogue: phi = (rs*mu - GP + (GP - p*rs)/h2) / N ----
#pragma unroll
    for (int k = 0; k < 8; ++k) {
        int idx = tid + k * 256;           // 0..2047
        int r = idx >> 7;
        int d = idx & 127;
        int i = ibase + r;
        float rsum = rsS[r];
        float gp   = accS[idx];
        float p    = particles[i * D + d];
        float val  = (rsum * mu[d] - gp + (gp - p * rsum) * invh2) * invN;
        out[i * D + d] = val;
    }
}

extern "C" void kernel_launch(void* const* d_in, const int* in_sizes, int n_in,
                              void* d_out, int out_size, void* d_ws, size_t ws_size,
                              hipStream_t stream) {
    const float* particles = (const float*)d_in[0];
    const float* mu        = (const float*)d_in[1];
    const float* bw        = (const float*)d_in[2];
    float* out = (float*)d_out;

    char* ws = (char*)d_ws;
    float*  sqv  = (float*)ws;                          // 4096 * 4 B
    __bf16* Phi  = (__bf16*)(ws + 16384);               // 1 MB
    __bf16* Plo  = Phi  + (size_t)N * D;                // 1 MB
    __bf16* PThi = Plo  + (size_t)N * D;                // 1 MB
    __bf16* PTlo = PThi + (size_t)N * D;                // 1 MB

    svgd_prep<<<N, D, 0, stream>>>(particles, sqv, Phi, Plo);
    dim3 tgrid(N / 32, D / 32);
    dim3 tblk(32, 8);
    svgd_transpose<<<tgrid, tblk, 0, stream>>>(Phi, Plo, PThi, PTlo);
    svgd_main<<<N / 16, 256, 0, stream>>>(particles, mu, bw, sqv, Phi, Plo, PThi, PTlo, out);
}
